// MHFSpectralConv2D_38199439131360
// MI455X (gfx1250) — matrix operand order repaired and checked by reading, not compiled, not run on hardware
//
#include <hip/hip_runtime.h>

// MI455X / gfx1250 — wave32, WMMA f16->f32 path.
// Pipeline: partial-DFT GEMMs via V_WMMA_F32_16X16X32_F16 (f32 accum),
// f16 twiddle tables + intermediates, LDS-staged B matrices.
// LDS staging uses GLOBAL_LOAD_ASYNC_TO_LDS_B128 (ASYNCcnt) when available.

typedef __attribute__((ext_vector_type(16))) _Float16 v16h;
typedef __attribute__((ext_vector_type(8)))  _Float16 v8h;
typedef __attribute__((ext_vector_type(8)))  float    v8f;
typedef __attribute__((ext_vector_type(4)))  float    v4f;

#define PI_F 3.14159265358979323846f

#if __has_builtin(__builtin_amdgcn_global_load_async_to_lds_b128) && \
    __has_builtin(__builtin_amdgcn_s_wait_asynccnt)
#define USE_ASYNC_LDS 1
typedef __attribute__((ext_vector_type(4))) int v4i;
typedef __attribute__((address_space(1))) v4i g_v4i;   // global int4
typedef __attribute__((address_space(3))) v4i l_v4i;   // LDS int4
#else
#define USE_ASYNC_LDS 0
#endif

static __device__ __forceinline__ v8f wmma16(v16h a, v16h b, v8f c) {
  // D = A(16x32 f16) * B(32x16 f16) + C(16x16 f32)
  return __builtin_amdgcn_wmma_f32_16x16x32_f16(false, a, false, b, (short)0, c,
                                                false, false);
}

static __device__ __forceinline__ v16h mk_a(const _Float16* p) {
  // lane's A fragment: 16 contiguous halves at p (two runs: p and p+8)
  v8h lo = *(const v8h*)(p);
  v8h hi = *(const v8h*)(p + 8);
  return __builtin_shufflevector(lo, hi, 0,1,2,3,4,5,6,7,8,9,10,11,12,13,14,15);
}

// Stage n16*16 bytes of twiddle table into LDS (block-cooperative), then barrier.
static __device__ __forceinline__ void stage_lds(_Float16* sB,
                                                 const _Float16* gsrc,
                                                 int n16, int tid) {
#if USE_ASYNC_LDS
  for (int i = tid; i < n16; i += 256) {
    __builtin_amdgcn_global_load_async_to_lds_b128(
        (g_v4i*)(uintptr_t)((const char*)gsrc + (size_t)i * 16),
        (l_v4i*)(unsigned)(uintptr_t)((char*)sB + (size_t)i * 16),
        0, 0);
  }
  __builtin_amdgcn_s_wait_asynccnt(0);  // drain this wave's async copies
#else
  const uint4* src = (const uint4*)gsrc;
  uint4* dst = (uint4*)sB;
  for (int i = tid; i < n16; i += 256) dst[i] = src[i];
#endif
  __syncthreads();
}

// ---------------------------------------------------------------- tables ----
// F1 [48 cols][256]  : col 2ky+p -> p? -sin : cos (2*pi*ky*w/256)/16 ; cols>=34 zero
// F2 [64 cols][256]  : col 2kx+p -> p? sin : cos  (2*pi*kx*h/256)/16
// F3 [512 cols][64]  : col 2h+p, row 2kx+q -> complex e^{+i th}/16 expansion
// F4 [256 cols][64]  : col w, row 2ky+q -> hermitian irfft weights /16 ; rows>=34 zero
__global__ __launch_bounds__(256) void k0_tables(_Float16* __restrict__ F1,
                                                 _Float16* __restrict__ F2,
                                                 _Float16* __restrict__ F3,
                                                 _Float16* __restrict__ F4) {
  int i = blockIdx.x * 256 + threadIdx.x;
  const float s  = 1.0f / 16.0f;
  const float w0 = 2.0f * PI_F / 256.0f;
  if (i < 48 * 256) {
    int n = i >> 8, w = i & 255;
    float v = 0.f;
    if (n < 34) {
      int ky = n >> 1;
      float th = w0 * (float)(ky * w);
      v = (n & 1) ? -__sinf(th) * s : __cosf(th) * s;
    }
    F1[i] = (_Float16)v;
  }
  if (i < 64 * 256) {
    int j = i >> 8, h = i & 255;
    int kx = j >> 1;
    float th = w0 * (float)(kx * h);
    F2[i] = (_Float16)((((j & 1) ? __sinf(th) : __cosf(th))) * s);
  }
  if (i < 512 * 64) {
    int n2 = i >> 6, k = i & 63;
    int h = n2 >> 1, p = n2 & 1, kx = k >> 1, q = k & 1;
    float th = w0 * (float)(kx * h);
    float v = (p == 0) ? ((q == 0) ? __cosf(th) : -__sinf(th))
                       : ((q == 0) ? __sinf(th) :  __cosf(th));
    F3[i] = (_Float16)(v * s);
  }
  if (i < 256 * 64) {
    int w = i >> 6, k = i & 63;
    float v = 0.f;
    if (k < 34) {
      int ky = k >> 1;
      float a  = (ky == 0) ? 1.f : 2.f;
      float th = w0 * (float)(ky * w);
      v = (((k & 1) ? -__sinf(th) : __cosf(th))) * a * s;
    }
    F4[i] = (_Float16)v;
  }
}

// ------------------------------------------------ K1: forward W-DFT GEMM ----
// x[131072,256] (f32, rows = bc*256+h) x F1[256,48] -> T1t[bc][n(36)][h] f16
__global__ __launch_bounds__(256) void k1_fwd_w(const float* __restrict__ x,
                                                const _Float16* __restrict__ F1,
                                                _Float16* __restrict__ T1) {
  __shared__ _Float16 sB[48 * 256];  // 24KB twiddles, col-major
  int tid = threadIdx.x;
  stage_lds(sB, F1, 1536, tid);
  int wave = tid >> 5, lane = tid & 31;
  int mt  = blockIdx.x * 8 + wave;     // M tile
  int row = mt * 16 + (lane & 15);     // A row = bc*256 + h
  const float* ap = x + (size_t)row * 256;
  int selA = (lane < 16) ? 0 : 16;
  int selB = (lane < 16) ? 0 : 16;
  int nl = lane & 15;
  v8f acc0 = {}, acc1 = {}, acc2 = {};
  for (int kt = 0; kt < 8; ++kt) {
    const float* p0 = ap + kt * 32 + selA;
    __builtin_prefetch(p0 + 64, 0, 1);          // global_prefetch K slab two iterations ahead
    v4f u0 = *(const v4f*)(p0);
    v4f u1 = *(const v4f*)(p0 + 4);
    v4f u2 = *(const v4f*)(p0 + 8);
    v4f u3 = *(const v4f*)(p0 + 12);
    v16h a;
    a[0]=(_Float16)u0[0]; a[1]=(_Float16)u0[1]; a[2]=(_Float16)u0[2]; a[3]=(_Float16)u0[3];
    a[4]=(_Float16)u1[0]; a[5]=(_Float16)u1[1]; a[6]=(_Float16)u1[2]; a[7]=(_Float16)u1[3];
    a[8]=(_Float16)u2[0]; a[9]=(_Float16)u2[1]; a[10]=(_Float16)u2[2]; a[11]=(_Float16)u2[3];
    a[12]=(_Float16)u3[0]; a[13]=(_Float16)u3[1]; a[14]=(_Float16)u3[2]; a[15]=(_Float16)u3[3];
    int bofs = kt * 32 + selB;
    v16h b0 = *(const v16h*)&sB[(0 * 16 + nl) * 256 + bofs];
    v16h b1 = *(const v16h*)&sB[(1 * 16 + nl) * 256 + bofs];
    v16h b2 = *(const v16h*)&sB[(2 * 16 + nl) * 256 + bofs];
    acc0 = wmma16(a, b0, acc0);
    acc1 = wmma16(a, b1, acc1);
    acc2 = wmma16(a, b2, acc2);
  }
  // D: lane holds N=nl, M = Mb..Mb+7 -> 8 consecutive h -> one 16B f16 store
  int Mb = mt * 16 + ((lane >> 4) << 3);
  int bc = Mb >> 8, h0 = Mb & 255;
  v8f accs[3] = {acc0, acc1, acc2};
#pragma unroll
  for (int nt = 0; nt < 3; ++nt) {
    int n = nt * 16 + nl;
    if (n < 36) {
      v8h o;
#pragma unroll
      for (int r = 0; r < 8; ++r) o[r] = (_Float16)accs[nt][r];
      *(v8h*)&T1[((size_t)(bc * 36 + n) << 8) + h0] = o;
    }
  }
}

// ------------------------------------------------ K2: forward H-DFT GEMM ----
// T1t rows [18432,256] f16 x F2[256,64] -> P[18432,64] f32 (cos/sin products)
__global__ __launch_bounds__(256) void k2_fwd_h(const _Float16* __restrict__ T1,
                                                const _Float16* __restrict__ F2,
                                                float* __restrict__ P) {
  __shared__ _Float16 sB[64 * 256];  // 32KB
  int tid = threadIdx.x;
  stage_lds(sB, F2, 2048, tid);
  int wave = tid >> 5, lane = tid & 31;
  int mt = blockIdx.x * 8 + wave;
  int arow = mt * 16 + (lane & 15);
  const _Float16* ap = T1 + (size_t)arow * 256;
  int selA = (lane < 16) ? 0 : 16;
  int selB = (lane < 16) ? 0 : 16;
  int nl = lane & 15;
  v8f acc[4] = {};
  for (int kt = 0; kt < 8; ++kt) {
    v16h a = mk_a(ap + kt * 32 + selA);
    int bofs = kt * 32 + selB;
#pragma unroll
    for (int nt = 0; nt < 4; ++nt) {
      v16h b = *(const v16h*)&sB[(nt * 16 + nl) * 256 + bofs];
      acc[nt] = wmma16(a, b, acc[nt]);
    }
  }
  int Mb = mt * 16 + ((lane >> 4) << 3);
#pragma unroll
  for (int nt = 0; nt < 4; ++nt) {
    int n = nt * 16 + nl;
#pragma unroll
    for (int r = 0; r < 8; ++r) P[(size_t)(Mb + r) * 64 + n] = acc[nt][r];
  }
}

// ------------------------------- K3: per-mode complex channel mix (VALU) ----
// Combine P -> Xf, multiply by weights, emit Yc[bo*17+ky][64] f16 (re/im by kx)
__global__ __launch_bounds__(256) void k3_mix(const float* __restrict__ P,
                                              const float* __restrict__ Wr,
                                              const float* __restrict__ Wi,
                                              _Float16* __restrict__ Yc) {
  __shared__ float sWr[64 * 64];
  __shared__ float sWi[64 * 64];
  __shared__ float sXr[8 * 64];
  __shared__ float sXi[8 * 64];
  int kx = blockIdx.x / 17;
  int ky = blockIdx.x - kx * 17;
  int tid = threadIdx.x;
  for (int i = tid; i < 4096; i += 256) {
    int c = i >> 6, o = i & 63;
    int hh = o >> 4, oo = o & 15;
    size_t widx = ((((size_t)hh * 64 + c) * 16 + oo) * 32 + kx) * 17 + ky;
    sWr[i] = Wr[widx];
    sWi[i] = Wi[widx];
  }
  for (int i = tid; i < 512; i += 256) {
    int b = i >> 6, c = i & 63;
    size_t base = ((size_t)(b * 64 + c) * 36 + 2 * ky) * 64 + 2 * kx;
    float p00 = P[base], p01 = P[base + 1], p10 = P[base + 64], p11 = P[base + 65];
    sXr[i] = p00 + p11;   // re = cos*re + sin*im
    sXi[i] = p10 - p01;   // im = cos*im - sin*re
  }
  __syncthreads();
  int o = tid & 63, b0 = tid >> 6;
  for (int bb = b0; bb < 8; bb += 4) {
    float ar = 0.f, ai = 0.f;
#pragma unroll 4
    for (int c = 0; c < 64; ++c) {
      float xr = sXr[bb * 64 + c], xi = sXi[bb * 64 + c];
      float wr = sWr[c * 64 + o],  wi = sWi[c * 64 + o];
      ar = fmaf(xr, wr, ar); ar = fmaf(-xi, wi, ar);
      ai = fmaf(xr, wi, ai); ai = fmaf( xi, wr, ai);
    }
    size_t row = (size_t)(bb * 64 + o) * 17 + ky;
    _Float16* yp = Yc + row * 64 + 2 * kx;
    yp[0] = (_Float16)ar;
    yp[1] = (_Float16)ai;
  }
}

// ------------------------------------------------ K4: inverse H-DFT GEMM ----
// Yc[8704,64] f16 x F3[64,512] -> T3[bo*256+h][64] f16 (col 2ky+p), scatter D
__global__ __launch_bounds__(256) void k4_inv_h(const _Float16* __restrict__ Yc,
                                                const _Float16* __restrict__ F3,
                                                _Float16* __restrict__ T3) {
  __shared__ _Float16 sB[512 * 64];  // 64KB
  int tid = threadIdx.x;
  stage_lds(sB, F3, 4096, tid);
  int wave = tid >> 5, lane = tid & 31;
  int mt = blockIdx.x * 8 + wave;          // 0..543
  int arow = mt * 16 + (lane & 15);        // bo*17+ky
  const _Float16* ap = Yc + (size_t)arow * 64;
  int selA = (lane < 16) ? 0 : 16;
  int selB = (lane < 16) ? 0 : 16;
  int nl = lane & 15;
  v16h a0 = mk_a(ap + selA);
  v16h a1 = mk_a(ap + 32 + selA);
  int Mb = mt * 16 + ((lane >> 4) << 3);
  for (int nt = 0; nt < 32; ++nt) {
    v8f acc = {};
    const _Float16* bp = &sB[(nt * 16 + nl) * 64];
    v16h b0 = *(const v16h*)(bp + selB);
    v16h b1 = *(const v16h*)(bp + 32 + selB);
    acc = wmma16(a0, b0, acc);
    acc = wmma16(a1, b1, acc);
    int N = nt * 16 + nl;
    int h = N >> 1, p = N & 1;
#pragma unroll
    for (int r = 0; r < 8; ++r) {
      int Mg = Mb + r;
      int bo = Mg / 17;
      int ky = Mg - bo * 17;
      T3[(size_t)((bo << 8) + h) * 64 + 2 * ky + p] = (_Float16)acc[r];
    }
  }
}

// ------------------------------------- K5: inverse W (irfft) GEMM + bias ----
// T3[131072,64] f16 x F4[64,256] -> out[131072,256] f32
__global__ __launch_bounds__(256) void k5_inv_w(const _Float16* __restrict__ T3,
                                                const _Float16* __restrict__ F4,
                                                const float* __restrict__ bias,
                                                float* __restrict__ out) {
  __shared__ _Float16 sB[256 * 64];  // 32KB
  int tid = threadIdx.x;
  stage_lds(sB, F4, 2048, tid);
  int wave = tid >> 5, lane = tid & 31;
  int mt = blockIdx.x * 8 + wave;          // 0..8191
  int arow = mt * 16 + (lane & 15);        // bo*256 + h
  const _Float16* ap = T3 + (size_t)arow * 64;
  __builtin_prefetch(ap, 0, 1);
  int selA = (lane < 16) ? 0 : 16;
  int selB = (lane < 16) ? 0 : 16;
  int nl = lane & 15;
  v16h a0 = mk_a(ap + selA);
  v16h a1 = mk_a(ap + 32 + selA);
  int Mb = mt * 16 + ((lane >> 4) << 3);
  float bv = bias[(Mb >> 8) & 63];         // per-channel bias, uniform per tile
  for (int nt = 0; nt < 16; ++nt) {
    v8f acc = {};
    const _Float16* bp = &sB[(nt * 16 + nl) * 64];
    v16h b0 = *(const v16h*)(bp + selB);
    v16h b1 = *(const v16h*)(bp + 32 + selB);
    acc = wmma16(a0, b0, acc);
    acc = wmma16(a1, b1, acc);
    int w = nt * 16 + nl;
#pragma unroll
    for (int r = 0; r < 8; ++r)
      out[(size_t)(Mb + r) * 256 + w] = acc[r] + bv;
  }
}

// -------------------------------------------------------------- launcher ----
extern "C" void kernel_launch(void* const* d_in, const int* in_sizes, int n_in,
                              void* d_out, int out_size, void* d_ws, size_t ws_size,
                              hipStream_t stream) {
  const float* x    = (const float*)d_in[0];
  const float* wr   = (const float*)d_in[1];
  const float* wi   = (const float*)d_in[2];
  const float* bias = (const float*)d_in[3];
  char* ws = (char*)d_ws;
  // workspace map (bytes), total ~32.2 MB
  _Float16* F1 = (_Float16*)(ws + 0);          //  24576
  _Float16* F2 = (_Float16*)(ws + 24576);      //  32768
  _Float16* F3 = (_Float16*)(ws + 57344);      //  65536
  _Float16* F4 = (_Float16*)(ws + 122880);     //  32768
  _Float16* T1 = (_Float16*)(ws + 155648);     //  9437184  [512][36][256] f16
  float*    P  = (float*)   (ws + 9592832);    //  4718592  [18432][64] f32
  _Float16* Yc = (_Float16*)(ws + 14311424);   //  1114112  [8704][64] f16
  _Float16* T3 = (_Float16*)(ws + 15425536);   // 16777216  [131072][64] f16
  if (ws_size < (size_t)32202752) return;
  float* out = (float*)d_out;

  hipLaunchKernelGGL(k0_tables, dim3(128),  dim3(256), 0, stream, F1, F2, F3, F4);
  hipLaunchKernelGGL(k1_fwd_w,  dim3(1024), dim3(256), 0, stream, x, F1, T1);
  hipLaunchKernelGGL(k2_fwd_h,  dim3(144),  dim3(256), 0, stream, T1, F2, P);
  hipLaunchKernelGGL(k3_mix,    dim3(544),  dim3(256), 0, stream, P, wr, wi, Yc);
  hipLaunchKernelGGL(k4_inv_h,  dim3(68),   dim3(256), 0, stream, Yc, F3, T3);
  hipLaunchKernelGGL(k5_inv_w,  dim3(1024), dim3(256), 0, stream, T3, F4, bias, out);
}
